// HeadParallelDilatedAttentionOptimized_23029614641316
// MI455X (gfx1250) — matrix operand order repaired and checked by reading, not compiled, not run on hardware
//
#include <hip/hip_runtime.h>
#include <stdint.h>

typedef __attribute__((ext_vector_type(16))) __bf16 v16bf;
typedef __attribute__((ext_vector_type(8)))  float  v8f;

union Frag {
  v16bf    bf;
  uint32_t u[8];
  uint4    q4[2];
};

__device__ __forceinline__ uint16_t f2bf(float x) {
  uint32_t u = __float_as_uint(x);
  u += 0x7FFFu + ((u >> 16) & 1u);       // round-to-nearest-even
  return (uint16_t)(u >> 16);
}
__device__ __forceinline__ uint32_t pk2(float lo, float hi) {
  return (uint32_t)f2bf(lo) | ((uint32_t)f2bf(hi) << 16);
}
__device__ __forceinline__ Frag ldsFrag(const uint16_t* p) {
  Frag f;
  f.q4[0] = *(const uint4*)(p);
  f.q4[1] = *(const uint4*)(p + 8);
  return f;
}
__device__ __forceinline__ v8f vzero() {
  v8f z;
  #pragma unroll
  for (int i = 0; i < 8; ++i) z[i] = 0.0f;
  return z;
}
__device__ __forceinline__ float rmax16(float v) {
  #pragma unroll
  for (int m = 8; m >= 1; m >>= 1) v = fmaxf(v, __shfl_xor(v, m, 16));
  return v;
}
__device__ __forceinline__ float rsum16(float v) {
  #pragma unroll
  for (int m = 8; m >= 1; m >>= 1) v += __shfl_xor(v, m, 16);
  return v;
}

#define SEQ 4096
#define NH  12
#define HD  64
#define KC  64    // keys per iteration
#define KT_S 72   // sKt row stride (bf16): rows = 64 keys, cols = 64 dims (+8 pad)
#define V_S  72   // sV  row stride (bf16): rows = 64 dims, cols = 64 keys (+8 pad)
#define P_S  72   // sP  row stride (bf16): 16 rows/wave,  cols = 64 keys (+8 pad)

struct TrueT  { static constexpr bool value = true;  };
struct FalseT { static constexpr bool value = false; };

__global__ __launch_bounds__(256)
void dilated_attn_fwd(const float* __restrict__ Q, const float* __restrict__ K,
                      const float* __restrict__ V, const int* __restrict__ isCausalPtr,
                      float* __restrict__ O) {
  __shared__ __align__(16) uint16_t sKt[64 * KT_S];     // K   : [key][dim]
  __shared__ __align__(16) uint16_t sV [64 * V_S];      // V^T : [dim][key]
  __shared__ __align__(16) uint16_t sP [8 * 16 * P_S];  // per-wave P transpose buffer

  const int tid  = threadIdx.x;
  const int lane = tid & 31;
  const int wave = tid >> 5;
  const int hi   = (lane >> 4) & 1;
  const int ln   = lane & 15;

  const int blk = blockIdx.x;
  const int qt  = blk & 31;      // 32 query tiles of 128 rows
  const int bh  = blk >> 5;      // 0..23
  const int h   = bh % NH;
  const int b   = bh / NH;
  const int q0  = qt * 128;
  const int group = h >> 2;          // 4 heads per group
  const int L     = 1024 << group;   // 1024 / 2048 / 4096
  const int seg   = q0 / L;
  const int kbase = seg * L;

  const int  causal  = isCausalPtr[0];
  const bool maskSeg = (causal != 0) && (group == 0) && (seg == 0);
  int kend = kbase + L;
  if (maskSeg) kend = min(kend, q0 + 128);

  const int q0w = q0 + wave * 16;
  const int myRow = q0w + ln;

  // ---- Q A-fragments (contiguous k order used here, same as B: lane<16 -> K{0..15}, lane>=16 -> K{16..31}) ----
  const float* qrow = Q + ((size_t)((b * SEQ + myRow) * NH + h)) * HD;
  Frag aq0, aq1;
  #pragma unroll
  for (int c = 0; c < 2; ++c) {
    Frag a;
    const int base = c * 32;
    const int g0 = base + hi * 16;
    float4 f0 = *(const float4*)(qrow + g0);
    float4 f1 = *(const float4*)(qrow + g0 + 4);
    a.u[0] = pk2(f0.x, f0.y); a.u[1] = pk2(f0.z, f0.w);
    a.u[2] = pk2(f1.x, f1.y); a.u[3] = pk2(f1.z, f1.w);
    const int g1 = base + hi * 16 + 8;
    float4 f2 = *(const float4*)(qrow + g1);
    float4 f3 = *(const float4*)(qrow + g1 + 4);
    a.u[4] = pk2(f2.x, f2.y); a.u[5] = pk2(f2.z, f2.w);
    a.u[6] = pk2(f3.x, f3.y); a.u[7] = pk2(f3.z, f3.w);
    if (c == 0) aq0 = a; else aq1 = a;
  }

  v8f o0 = vzero(), o1 = vzero(), o2 = vzero(), o3 = vzero();
  float m_r[8], l_r[8];
  #pragma unroll
  for (int r = 0; r < 8; ++r) { m_r[r] = -1e30f; l_r[r] = 0.0f; }

  const float cA = 0.125f * 1.44269504088896340736f;  // (1/sqrt(64)) * log2(e)

  const int key = tid >> 2;          // 0..63 : key row this thread stages
  const int db  = (tid & 3) * 16;    // dim base (16 floats per thread)

  uint16_t* pw = &sP[(wave * 16) * P_S];

  auto runLoop = [&](auto maskc) {
    constexpr bool MASK = decltype(maskc)::value;

    for (int k0 = kbase; k0 < kend; k0 += KC) {
      __syncthreads();  // previous iteration's LDS consumers are done

      // ---- cooperative stage: K row-major, V transposed (f32 -> bf16) ----
      {
        const size_t gidx = ((size_t)((b * SEQ + (k0 + key)) * NH + h)) * HD + db;
        const float* kp = K + gidx;
        #pragma unroll
        for (int j = 0; j < 2; ++j) {
          float4 b0 = *(const float4*)(kp + 8 * j);
          float4 b1 = *(const float4*)(kp + 8 * j + 4);
          uint4 w;
          w.x = pk2(b0.x, b0.y); w.y = pk2(b0.z, b0.w);
          w.z = pk2(b1.x, b1.y); w.w = pk2(b1.z, b1.w);
          *(uint4*)&sKt[key * KT_S + db + 8 * j] = w;
        }
        const float* vp = V + gidx;
        #pragma unroll
        for (int j = 0; j < 4; ++j) {
          float4 a0 = *(const float4*)(vp + 4 * j);
          sV[(db + 4 * j + 0) * V_S + key] = f2bf(a0.x);
          sV[(db + 4 * j + 1) * V_S + key] = f2bf(a0.y);
          sV[(db + 4 * j + 2) * V_S + key] = f2bf(a0.z);
          sV[(db + 4 * j + 3) * V_S + key] = f2bf(a0.w);
        }
        if (k0 + KC < kend) {                      // prefetch next tile into cache
          __builtin_prefetch(kp + KC * NH * HD, 0, 1);
          __builtin_prefetch(vp + KC * NH * HD, 0, 1);
        }
      }
      __syncthreads();

      // ---- S = Q * K^T : 8x v_wmma_f32_16x16x32_bf16 (4 key subtiles x 2 dim chunks) ----
      v8f s[4];
      #pragma unroll
      for (int t = 0; t < 4; ++t) {
        Frag bk = ldsFrag(&sKt[(16 * t + ln) * KT_S + hi * 16]);          // dims 0..31
        v8f acc = vzero();
        acc = __builtin_amdgcn_wmma_f32_16x16x32_bf16(false, aq0.bf, false, bk.bf, (short)0, acc, false, false);
        bk = ldsFrag(&sKt[(16 * t + ln) * KT_S + 32 + hi * 16]);        // dims 32..63
        acc = __builtin_amdgcn_wmma_f32_16x16x32_bf16(false, aq1.bf, false, bk.bf, (short)0, acc, false, false);
        s[t] = acc;
      }

      // ---- online softmax (log2 domain); C-layout: row = r + 8*hi, col = 16*t + ln ----
      #pragma unroll
      for (int r = 0; r < 8; ++r) {
        float y[4];
        #pragma unroll
        for (int t = 0; t < 4; ++t) {
          y[t] = s[t][r] * cA;
          if (MASK) {
            const int qr = q0w + r + 8 * hi;
            y[t] = ((k0 + 16 * t + ln) > qr) ? -1e30f : y[t];
          }
        }
        float mx = fmaxf(fmaxf(y[0], y[1]), fmaxf(y[2], y[3]));
        mx = rmax16(mx);
        const float mnew  = fmaxf(m_r[r], mx);
        const float alpha = exp2f(m_r[r] - mnew);
        float p[4], ps = 0.0f;
        #pragma unroll
        for (int t = 0; t < 4; ++t) { p[t] = exp2f(y[t] - mnew); ps += p[t]; }
        const float rs = rsum16(ps);
        l_r[r] = l_r[r] * alpha + rs;
        m_r[r] = mnew;
        o0[r] *= alpha; o1[r] *= alpha; o2[r] *= alpha; o3[r] *= alpha;
        #pragma unroll
        for (int t = 0; t < 4; ++t)
          pw[(r + 8 * hi) * P_S + 16 * t + ln] = f2bf(p[t]);
      }

      // cross-lane LDS dependency (P write -> P read within the wave)
      asm volatile("s_wait_dscnt 0" ::: "memory");

      // ---- O += P * V : 8x v_wmma_f32_16x16x32_bf16 (2 key chunks x 4 dim subtiles) ----
      #pragma unroll
      for (int kc = 0; kc < 2; ++kc) {
        Frag ap;
        const uint16_t* pr = &pw[ln * P_S + kc * 32];
        ap.q4[0] = *(const uint4*)(pr + hi * 16);
        ap.q4[1] = *(const uint4*)(pr + hi * 16 + 8);
        const int krow = kc * 32 + hi * 16;
        Frag bv = ldsFrag(&sV[(0 + ln) * V_S + krow]);
        o0 = __builtin_amdgcn_wmma_f32_16x16x32_bf16(false, ap.bf, false, bv.bf, (short)0, o0, false, false);
        bv = ldsFrag(&sV[(16 + ln) * V_S + krow]);
        o1 = __builtin_amdgcn_wmma_f32_16x16x32_bf16(false, ap.bf, false, bv.bf, (short)0, o1, false, false);
        bv = ldsFrag(&sV[(32 + ln) * V_S + krow]);
        o2 = __builtin_amdgcn_wmma_f32_16x16x32_bf16(false, ap.bf, false, bv.bf, (short)0, o2, false, false);
        bv = ldsFrag(&sV[(48 + ln) * V_S + krow]);
        o3 = __builtin_amdgcn_wmma_f32_16x16x32_bf16(false, ap.bf, false, bv.bf, (short)0, o3, false, false);
      }
    }
  };

  if (maskSeg) runLoop(TrueT{}); else runLoop(FalseT{});

  // ---- finalize: divide by l, store f32 ----
  #pragma unroll
  for (int r = 0; r < 8; ++r) {
    const float inv = 1.0f / l_r[r];
    const int row = q0w + r + 8 * hi;
    float* op = O + ((size_t)((b * SEQ + row) * NH + h)) * HD + ln;
    op[0]  = o0[r] * inv;
    op[16] = o1[r] * inv;
    op[32] = o2[r] * inv;
    op[48] = o3[r] * inv;
  }
}

extern "C" void kernel_launch(void* const* d_in, const int* in_sizes, int n_in,
                              void* d_out, int out_size, void* d_ws, size_t ws_size,
                              hipStream_t stream) {
  (void)in_sizes; (void)n_in; (void)out_size; (void)d_ws; (void)ws_size;
  const float* q = (const float*)d_in[0];
  const float* k = (const float*)d_in[1];
  const float* v = (const float*)d_in[2];
  const int*   c = (const int*)d_in[3];
  float* out = (float*)d_out;
  // grid: 2 batches * 12 heads * 32 query tiles (128 rows) = 768 blocks
  dim3 grid(768), block(256);
  hipLaunchKernelGGL(dilated_attn_fwd, grid, block, 0, stream, q, k, v, c, out);
}
